// PhysMamba1D_44392781971764
// MI455X (gfx1250) — matrix operand order repaired and checked by reading, not compiled, not run on hardware
//
#include <hip/hip_runtime.h>
#include <hip/hip_bf16.h>
#include <math.h>

// ---------------------------------------------------------------------------
// PhysMamba1D for MI455X (gfx1250): WMMA f16 GEMMs + wave32-mapped SSM scan.
// K is a template parameter so the K-loop fully unrolls -> unconditional
// float4 (b128) fragment loads and a straight-line chain of v_wmma ops.
// ---------------------------------------------------------------------------

#define D_MODEL 128
#define D_STATE 32
#define D_CONV  8
#define N_BLOCKS 6
#define K_PRE 7
#define D_INNER 256            // 2*D_MODEL
#define DT_RANK 8
#define BB 2
#define LL 1024
#define M_TOK (BB * LL)        // 2048 token rows

typedef __attribute__((ext_vector_type(16))) _Float16 v16h;
typedef __attribute__((ext_vector_type(8)))  float    v8f;

// ---------------------------------------------------------------------------
// WMMA GEMM: C[M,N] = A[M,K] * W[N,K]^T (+bias[n]) (+resid[m,n])
//   K: compile-time, multiple of 32. lda/ldc: row strides (multiples of 4).
//   revA/revC: fold the reference's [:, ::-1] into row indexing.
// One wave computes one 16x16 tile; block = 4 waves covering 16(M) x 64(N).
// Fragment layouts per CDNA5 ISA 7.12.2 (wave32):
//   A (16x32 f16): lane m = lidx; halves 0-7  -> K = k0 + 8*hsel + h
//                                 halves 8-15 -> K = k0 + 16 + 8*hsel + (h-8)
//   B (32x16 f16): lane n = lidx; same K mapping as A
//   C/D: VGPR r, lane l -> M = r + 8*hsel, N = lidx
// ---------------------------------------------------------------------------
template<int K>
__global__ void wmma_gemm_kernel(const float* __restrict__ A,
                                 const float* __restrict__ W,
                                 const float* __restrict__ bias,
                                 const float* __restrict__ resid,
                                 float* __restrict__ C,
                                 int M, int N, int lda, int ldc,
                                 int revA, int revC, int L)
{
    const int wave = threadIdx.x >> 5;
    const int lane = threadIdx.x & 31;
    const int lidx = lane & 15;
    const int hsel = lane >> 4;            // 0: lanes 0-15, 1: lanes 16-31

    const int m0 = blockIdx.y * 16;
    const int n0 = blockIdx.x * 64 + wave * 16;
    if (m0 >= M || n0 >= N) return;

    // A fragment source row (with optional L-reversal of the token index)
    int arow = m0 + lidx;
    if (revA) { int b = arow / L, l = arow % L; arow = b * L + (L - 1 - l); }
    const float* __restrict__ Arow = A + (size_t)arow * lda;

    // Clamp W row so loads are always in-bounds; stores are guarded below.
    int wrow = n0 + lidx;
    if (wrow > N - 1) wrow = N - 1;
    const float* __restrict__ Wrow = W + (size_t)wrow * K;

    v8f acc = {};

#pragma unroll
    for (int k0 = 0; k0 < K; k0 += 32) {
        if (k0 + 32 < K) {
            __builtin_prefetch(Arow + k0 + 32, 0, 1);   // global_prefetch_b8
            __builtin_prefetch(Wrow + k0 + 32, 0, 1);
        }
        // --- A fragment: two contiguous 8-float chunks per lane
        const float4 a0 = *(const float4*)(Arow + k0 + 8 * hsel);
        const float4 a1 = *(const float4*)(Arow + k0 + 8 * hsel + 4);
        const float4 a2 = *(const float4*)(Arow + k0 + 8 * hsel + 16);
        const float4 a3 = *(const float4*)(Arow + k0 + 8 * hsel + 20);
        // --- B fragment: two contiguous 8-float chunks per lane
        const float4 b0 = *(const float4*)(Wrow + k0 + 8 * hsel);
        const float4 b1 = *(const float4*)(Wrow + k0 + 8 * hsel + 4);
        const float4 b2 = *(const float4*)(Wrow + k0 + 8 * hsel + 16);
        const float4 b3 = *(const float4*)(Wrow + k0 + 8 * hsel + 20);

        v16h af, bf;
        af[0]  = (_Float16)a0.x; af[1]  = (_Float16)a0.y;
        af[2]  = (_Float16)a0.z; af[3]  = (_Float16)a0.w;
        af[4]  = (_Float16)a1.x; af[5]  = (_Float16)a1.y;
        af[6]  = (_Float16)a1.z; af[7]  = (_Float16)a1.w;
        af[8]  = (_Float16)a2.x; af[9]  = (_Float16)a2.y;
        af[10] = (_Float16)a2.z; af[11] = (_Float16)a2.w;
        af[12] = (_Float16)a3.x; af[13] = (_Float16)a3.y;
        af[14] = (_Float16)a3.z; af[15] = (_Float16)a3.w;

        bf[0]  = (_Float16)b0.x; bf[1]  = (_Float16)b0.y;
        bf[2]  = (_Float16)b0.z; bf[3]  = (_Float16)b0.w;
        bf[4]  = (_Float16)b1.x; bf[5]  = (_Float16)b1.y;
        bf[6]  = (_Float16)b1.z; bf[7]  = (_Float16)b1.w;
        bf[8]  = (_Float16)b2.x; bf[9]  = (_Float16)b2.y;
        bf[10] = (_Float16)b2.z; bf[11] = (_Float16)b2.w;
        bf[12] = (_Float16)b3.x; bf[13] = (_Float16)b3.y;
        bf[14] = (_Float16)b3.z; bf[15] = (_Float16)b3.w;

        acc = __builtin_amdgcn_wmma_f32_16x16x32_f16(
                  false, af, false, bf, (short)0, acc, false, false);
    }

    // --- store D: VGPR r, lane l -> M = m0 + r + 8*hsel, N = n0 + lidx
    const int nn = n0 + lidx;
    if (nn >= N) return;
    const float bv = bias ? bias[nn] : 0.0f;
#pragma unroll
    for (int r = 0; r < 8; ++r) {
        int mm = m0 + r + (hsel ? 8 : 0);
        int mme = mm;
        if (revC) { int b = mm / L, l = mm % L; mme = b * L + (L - 1 - l); }
        float v = acc[r] + bv;
        if (resid) v += resid[(size_t)mme * ldc + nn];
        C[(size_t)mme * ldc + nn] = v;
    }
}

// ---------------------------------------------------------------------------
// dt_proj: dtraw[t,d] = dt_b[d] + sum_{j<8} dbc[t,j] * dt_w[d,j]   (K=8: VALU)
// ---------------------------------------------------------------------------
__global__ void dtproj_kernel(const float* __restrict__ dbc,
                              const float* __restrict__ w,
                              const float* __restrict__ b,
                              float* __restrict__ dtraw)
{
    const int idx = blockIdx.x * blockDim.x + threadIdx.x;   // t*256 + d
    if (idx >= M_TOK * D_INNER) return;
    const int d = idx & (D_INNER - 1);
    const int t = idx >> 8;
    float s = b[d];
#pragma unroll
    for (int j = 0; j < DT_RANK; ++j)
        s += dbc[(size_t)t * (DT_RANK + 2 * D_STATE) + j] * w[d * DT_RANK + j];
    dtraw[idx] = s;
}

// ---------------------------------------------------------------------------
// Pre-conv (1 -> 128 channels, K=7, same padding) + exact GELU.
// ---------------------------------------------------------------------------
__global__ void preconv_gelu_kernel(const float* __restrict__ x,
                                    const float* __restrict__ w,
                                    const float* __restrict__ b,
                                    float* __restrict__ out)
{
    const int t = blockIdx.x;            // b*L + l
    const int bb = t >> 10, l = t & (LL - 1);
    const int c = threadIdx.x;
    float s = b[c];
#pragma unroll
    for (int k = 0; k < K_PRE; ++k) {
        int ll = l + k - (K_PRE / 2);
        if (ll >= 0 && ll < LL) s += x[bb * LL + ll] * w[c * K_PRE + k];
    }
    out[(size_t)t * D_MODEL + c] = 0.5f * s * (1.0f + erff(s * 0.70710678118f));
}

// ---------------------------------------------------------------------------
// LayerNorm over D_MODEL=128. One block (128 threads) per token row.
// ---------------------------------------------------------------------------
__global__ void layernorm_kernel(const float* __restrict__ h,
                                 const float* __restrict__ g,
                                 const float* __restrict__ b,
                                 float* __restrict__ xn)
{
    __shared__ float red[D_MODEL];
    const int row = blockIdx.x;
    const int c = threadIdx.x;
    const float v = h[(size_t)row * D_MODEL + c];
    red[c] = v;
    __syncthreads();
    for (int s = 64; s > 0; s >>= 1) { if (c < s) red[c] += red[c + s]; __syncthreads(); }
    const float mean = red[0] * (1.0f / D_MODEL);
    __syncthreads();
    const float d = v - mean;
    red[c] = d * d;
    __syncthreads();
    for (int s = 64; s > 0; s >>= 1) { if (c < s) red[c] += red[c + s]; __syncthreads(); }
    const float var = red[0] * (1.0f / D_MODEL);
    xn[(size_t)row * D_MODEL + c] = d * rsqrtf(var + 1e-5f) * g[c] + b[c];
}

// ---------------------------------------------------------------------------
// Depthwise causal conv (K=8) over xi = xz[..., :256], + bias + SiLU.
// ---------------------------------------------------------------------------
__global__ void dwconv_silu_kernel(const float* __restrict__ xz,
                                   const float* __restrict__ cw,
                                   const float* __restrict__ cb,
                                   float* __restrict__ xi)
{
    const int idx = blockIdx.x * blockDim.x + threadIdx.x;   // t*256 + d
    if (idx >= M_TOK * D_INNER) return;
    const int d = idx & (D_INNER - 1);
    const int t = idx >> 8;
    const int bb = t >> 10, l = t & (LL - 1);
    float s = cb[d];
#pragma unroll
    for (int k = 0; k < D_CONV; ++k) {
        int ll = l - (D_CONV - 1) + k;
        if (ll >= 0)
            s += xz[(size_t)(bb * LL + ll) * (2 * D_INNER) + d] * cw[d * D_CONV + k];
    }
    xi[idx] = s / (1.0f + expf(-s));                          // SiLU
}

// ---------------------------------------------------------------------------
// Selective scan: one wave32 per (batch, channel d); lane == state index n.
//   h_n <- exp(softplus(dt)*A_n)*h_n + softplus(dt)*xi*B_n
//   y_t  = sum_n h_n*C_n + xi*D_d ;  out = y * silu(z)
// Cross-lane sum via 5-step shfl_xor butterfly (wave32).
// ---------------------------------------------------------------------------
__global__ void scan_kernel(const float* __restrict__ dtraw,
                            const float* __restrict__ xi,
                            const float* __restrict__ dbc,
                            const float* __restrict__ xz,   // z = xz[..., 256:512]
                            const float* __restrict__ A_log,
                            const float* __restrict__ Dp,
                            float* __restrict__ ymul)
{
    const int w = (blockIdx.x * blockDim.x + threadIdx.x) >> 5;
    const int lane = threadIdx.x & 31;
    if (w >= BB * D_INNER) return;
    const int bb = w >> 8;
    const int d  = w & (D_INNER - 1);

    const float An = -expf(A_log[d * D_STATE + lane]);
    const float Dd = Dp[d];
    float h = 0.0f;

    for (int l = 0; l < LL; ++l) {
        const size_t t = (size_t)(bb * LL + l);
        const float dtr = dtraw[t * D_INNER + d];
        const float dt  = (dtr > 20.0f) ? dtr : log1pf(expf(dtr));
        const float xv  = xi[t * D_INNER + d];
        const float Bn  = dbc[t * (DT_RANK + 2 * D_STATE) + DT_RANK + lane];
        const float Cn  = dbc[t * (DT_RANK + 2 * D_STATE) + DT_RANK + D_STATE + lane];

        h = expf(dt * An) * h + dt * xv * Bn;
        float p = h * Cn;
#pragma unroll
        for (int s = 16; s > 0; s >>= 1) p += __shfl_xor(p, s, 32);

        if (lane == 0) {
            const float y  = p + xv * Dd;
            const float zv = xz[t * (2 * D_INNER) + D_INNER + d];
            ymul[t * D_INNER + d] = y * (zv / (1.0f + expf(-zv)));
        }
    }
}

// ---------------------------------------------------------------------------
// Squeeze-excite: mean pool over L, relu(fc1), sigmoid(fc2) -> attn[B,128].
// ---------------------------------------------------------------------------
__global__ void se_kernel(const float* __restrict__ h,
                          const float* __restrict__ w1, const float* __restrict__ b1,
                          const float* __restrict__ w2, const float* __restrict__ b2,
                          float* __restrict__ attn)
{
    __shared__ float pooled[D_MODEL];
    __shared__ float hid[D_MODEL / 4];
    const int c = threadIdx.x;
    for (int bb = 0; bb < BB; ++bb) {
        float s = 0.0f;
        for (int l = 0; l < LL; ++l) s += h[(size_t)(bb * LL + l) * D_MODEL + c];
        pooled[c] = s * (1.0f / LL);
        __syncthreads();
        if (c < D_MODEL / 4) {
            float a = b1[c];
            for (int k = 0; k < D_MODEL; ++k) a += pooled[k] * w1[c * D_MODEL + k];
            hid[c] = fmaxf(a, 0.0f);
        }
        __syncthreads();
        float a2 = b2[c];
        for (int j = 0; j < D_MODEL / 4; ++j) a2 += hid[j] * w2[c * (D_MODEL / 4) + j];
        attn[bb * D_MODEL + c] = 1.0f / (1.0f + expf(-a2));
        __syncthreads();
    }
}

// ---------------------------------------------------------------------------
// Final: out[b,l] = sum_c h[b,l,c]*attn[b,c]*out_w[c] + out_b
// ---------------------------------------------------------------------------
__global__ void final_kernel(const float* __restrict__ h,
                             const float* __restrict__ attn,
                             const float* __restrict__ ow,
                             const float* __restrict__ ob,
                             float* __restrict__ out)
{
    __shared__ float red[D_MODEL];
    const int t = blockIdx.x;
    const int bb = t >> 10;
    const int c = threadIdx.x;
    red[c] = h[(size_t)t * D_MODEL + c] * attn[bb * D_MODEL + c] * ow[c];
    __syncthreads();
    for (int s = 64; s > 0; s >>= 1) { if (c < s) red[c] += red[c + s]; __syncthreads(); }
    if (c == 0) out[t] = red[0] + ob[0];
}

// ---------------------------------------------------------------------------
// Host orchestration
// ---------------------------------------------------------------------------
static inline void launch_gemm(const float* A, const float* W, const float* bias,
                               const float* resid, float* C,
                               int M, int N, int K, int lda, int ldc,
                               int revA, int revC, hipStream_t stream)
{
    dim3 grid((N + 63) / 64, (M + 15) / 16);
    if (K == 128)
        wmma_gemm_kernel<128><<<grid, 128, 0, stream>>>(A, W, bias, resid, C,
                                                        M, N, lda, ldc, revA, revC, LL);
    else
        wmma_gemm_kernel<256><<<grid, 128, 0, stream>>>(A, W, bias, resid, C,
                                                        M, N, lda, ldc, revA, revC, LL);
}

extern "C" void kernel_launch(void* const* d_in, const int* in_sizes, int n_in,
                              void* d_out, int out_size, void* d_ws, size_t ws_size,
                              hipStream_t stream)
{
    const float* x       = (const float*)d_in[0];
    const float* pre_w   = (const float*)d_in[1];
    const float* pre_b   = (const float*)d_in[2];
    const float* ln_g    = (const float*)d_in[3];
    const float* ln_b    = (const float*)d_in[4];
    const float* in_w    = (const float*)d_in[5];
    const float* conv_w  = (const float*)d_in[6];
    const float* conv_b  = (const float*)d_in[7];
    const float* xproj_w = (const float*)d_in[8];
    const float* dt_w    = (const float*)d_in[9];
    const float* dt_b    = (const float*)d_in[10];
    const float* A_log   = (const float*)d_in[11];
    const float* Dp      = (const float*)d_in[12];
    const float* mo_w    = (const float*)d_in[13];
    const float* gate_w  = (const float*)d_in[14];
    const float* gate_b  = (const float*)d_in[15];
    const float* se_w1   = (const float*)d_in[16];
    const float* se_b1   = (const float*)d_in[17];
    const float* se_w2   = (const float*)d_in[18];
    const float* se_b2   = (const float*)d_in[19];
    const float* out_w   = (const float*)d_in[20];
    const float* out_b   = (const float*)d_in[21];
    float* out = (float*)d_out;

    // workspace layout (floats); all offsets are 16B-aligned
    float* ws = (float*)d_ws;
    float* h     = ws;                                   // 2048*128
    float* xn    = h     + (size_t)M_TOK * D_MODEL;      // 2048*128
    float* xz    = xn    + (size_t)M_TOK * D_MODEL;      // 2048*512
    float* xi    = xz    + (size_t)M_TOK * 2 * D_INNER;  // 2048*256
    float* dbc   = xi    + (size_t)M_TOK * D_INNER;      // 2048*72
    float* dtraw = dbc   + (size_t)M_TOK * (DT_RANK + 2 * D_STATE); // 2048*256
    float* ymul  = dtraw + (size_t)M_TOK * D_INNER;      // 2048*256
    float* fbw   = ymul  + (size_t)M_TOK * D_INNER;      // 2048*256
    float* attn  = fbw   + (size_t)M_TOK * D_INNER;      // 2*128
    (void)ws_size; (void)in_sizes; (void)n_in; (void)out_size;

    // 1) pre-conv + GELU -> h
    preconv_gelu_kernel<<<M_TOK, D_MODEL, 0, stream>>>(x, pre_w, pre_b, h);

    // 2) mamba blocks
    for (int i = 0; i < N_BLOCKS; ++i) {
        layernorm_kernel<<<M_TOK, D_MODEL, 0, stream>>>(
            h, ln_g + i * D_MODEL, ln_b + i * D_MODEL, xn);

        for (int dir = 0; dir < 2; ++dir) {
            const int p = i * 2 + dir;
            // in_proj: xz = xn(@rev) @ in_w^T   [2048 x 512 x 128]
            launch_gemm(xn, in_w + (size_t)p * (2 * D_INNER) * D_MODEL,
                        nullptr, nullptr, xz,
                        M_TOK, 2 * D_INNER, D_MODEL, D_MODEL, 2 * D_INNER,
                        /*revA=*/dir, /*revC=*/0, stream);
            // causal depthwise conv + SiLU -> xi
            dwconv_silu_kernel<<<(M_TOK * D_INNER + 255) / 256, 256, 0, stream>>>(
                xz, conv_w + (size_t)p * D_INNER * D_CONV,
                conv_b + (size_t)p * D_INNER, xi);
            // xproj: dbc = xi @ xproj_w^T   [2048 x 72 x 256]
            launch_gemm(xi, xproj_w + (size_t)p * (DT_RANK + 2 * D_STATE) * D_INNER,
                        nullptr, nullptr, dbc,
                        M_TOK, DT_RANK + 2 * D_STATE, D_INNER,
                        D_INNER, DT_RANK + 2 * D_STATE, 0, 0, stream);
            // dt_proj (K=8): plain VALU kernel
            dtproj_kernel<<<(M_TOK * D_INNER + 255) / 256, 256, 0, stream>>>(
                dbc, dt_w + (size_t)p * D_INNER * DT_RANK,
                dt_b + (size_t)p * D_INNER, dtraw);
            // selective scan (+ xi*D, * silu(z)) -> ymul
            scan_kernel<<<(BB * D_INNER * 32 + 255) / 256, 256, 0, stream>>>(
                dtraw, xi, dbc, xz,
                A_log + (size_t)p * D_INNER * D_STATE,
                Dp + (size_t)p * D_INNER, ymul);
            // out_proj -> fbw[:, dir*128 : dir*128+128], rows reversed for bw
            launch_gemm(ymul, mo_w + (size_t)p * D_MODEL * D_INNER,
                        nullptr, nullptr, fbw + dir * D_MODEL,
                        M_TOK, D_MODEL, D_INNER, D_INNER, 2 * D_MODEL,
                        /*revA=*/0, /*revC=*/dir, stream);
        }
        // gate + residual: h = h + fbw @ gate_w^T + gate_b   [2048 x 128 x 256]
        launch_gemm(fbw, gate_w + (size_t)i * D_MODEL * (2 * D_MODEL),
                    gate_b + i * D_MODEL, /*resid=*/h, h,
                    M_TOK, D_MODEL, 2 * D_MODEL, 2 * D_MODEL, D_MODEL,
                    0, 0, stream);
    }

    // 3) squeeze-excite + final projection
    se_kernel<<<1, D_MODEL, 0, stream>>>(h, se_w1, se_b1, se_w2, se_b2, attn);
    final_kernel<<<M_TOK, D_MODEL, 0, stream>>>(h, attn, out_w, out_b, out);
}
